// SelfAttn_47364899340976
// MI455X (gfx1250) — compile-verified, matrix operand order checked by reading, not run on hardware
//
#include <hip/hip_runtime.h>
#include <hip/hip_bf16.h>

// ---------------------------------------------------------------------------
// SAGAN self-attention block for MI455X (gfx1250, wave32, WMMA).
// Flash-attention formulation: never materialize the [N, M] attention matrix.
// All matmuls run through v_wmma_f32_16x16x32_bf16 (bf16 in, f32 accum).
// Softmax row reductions are done in A-layout (one row per lane-pair) so the
// whole online-softmax update needs only 2 shuffles per 32-key tile.
// ---------------------------------------------------------------------------

typedef __attribute__((ext_vector_type(16))) __bf16 v16bf;
typedef __attribute__((ext_vector_type(8)))  __bf16 v8bf;
typedef __attribute__((ext_vector_type(8)))  float  v8f;
typedef __attribute__((ext_vector_type(4)))  float  v4f;

#define B_  2
#define C_  64
#define H_  128
#define W_  128
#define N_  (H_ * W_)             // 16384 queries
#define M_  ((H_ / 2) * (W_ / 2)) // 4096 keys/values
#define D_  8                     // qk depth  (C/8)
#define DV_ 32                    // v depth   (C/2)

// round-to-nearest-even f32 -> bf16 without relying on __bf16 conversions
__device__ __forceinline__ __bf16 f2bf(float f) {
    unsigned u = __builtin_bit_cast(unsigned, f);
    unsigned r = (u + 0x7FFFu + ((u >> 16) & 1u)) >> 16;
    return __builtin_bit_cast(__bf16, (unsigned short)r);
}

// ---------------------------------------------------------------------------
// Projection kernel 1: theta[b, n, 0:8] = w_theta @ x[b, :, n]   (bf16 out)
// ---------------------------------------------------------------------------
__global__ void __launch_bounds__(256)
k_theta(const float* __restrict__ x, const float* __restrict__ w_theta,
        __bf16* __restrict__ theta) {
    __shared__ float wsh[D_ * C_];
    for (int i = threadIdx.x; i < D_ * C_; i += blockDim.x) wsh[i] = w_theta[i];
    __syncthreads();

    int idx = blockIdx.x * blockDim.x + threadIdx.x;   // over B*N (exact multiple)
    int b = idx >> 14;
    int p = idx & (N_ - 1);

    float acc[D_];
#pragma unroll
    for (int d = 0; d < D_; ++d) acc[d] = 0.f;

    const float* xb = x + (size_t)b * C_ * N_ + p;
    for (int c = 0; c < C_; ++c) {
        float xv = xb[(size_t)c * N_];
#pragma unroll
        for (int d = 0; d < D_; ++d) acc[d] = fmaf(wsh[d * C_ + c], xv, acc[d]);
    }
#pragma unroll
    for (int d = 0; d < D_; ++d) theta[(size_t)idx * D_ + d] = f2bf(acc[d]);
}

// ---------------------------------------------------------------------------
// Projection kernel 2: phi/g projections followed by 2x2 maxpool.
//   phi out: [B, M, 8]  (key-major, bf16)
//   g   out: [B, 32, M] (v-major, bf16 -> contiguous K for WMMA B operand)
// ---------------------------------------------------------------------------
__global__ void __launch_bounds__(256)
k_pool(const float* __restrict__ x, const float* __restrict__ w_phi,
       const float* __restrict__ w_g, __bf16* __restrict__ phi,
       __bf16* __restrict__ g) {
    __shared__ float wp[D_ * C_];
    __shared__ float wg[DV_ * C_];
    for (int i = threadIdx.x; i < D_ * C_; i += blockDim.x)  wp[i] = w_phi[i];
    for (int i = threadIdx.x; i < DV_ * C_; i += blockDim.x) wg[i] = w_g[i];
    __syncthreads();

    int idx = blockIdx.x * blockDim.x + threadIdx.x;   // over B*M (exact multiple)
    int b  = idx >> 12;
    int pm = idx & (M_ - 1);
    int i  = pm >> 6;           // H/2 = 64 rows
    int j  = pm & 63;           // W/2 = 64 cols

    float aphi[D_], ag[DV_];
#pragma unroll
    for (int d = 0; d < D_; ++d)  aphi[d] = -3.0e38f;
#pragma unroll
    for (int v = 0; v < DV_; ++v) ag[v]   = -3.0e38f;

    const float* xb = x + (size_t)b * C_ * N_;
#pragma unroll
    for (int dy = 0; dy < 2; ++dy) {
#pragma unroll
        for (int dx = 0; dx < 2; ++dx) {
            int n = (2 * i + dy) * W_ + (2 * j + dx);
            float pphi[D_], pg[DV_];
#pragma unroll
            for (int d = 0; d < D_; ++d)  pphi[d] = 0.f;
#pragma unroll
            for (int v = 0; v < DV_; ++v) pg[v]   = 0.f;
            for (int c = 0; c < C_; ++c) {
                float xv = xb[(size_t)c * N_ + n];
#pragma unroll
                for (int d = 0; d < D_; ++d)  pphi[d] = fmaf(wp[d * C_ + c], xv, pphi[d]);
#pragma unroll
                for (int v = 0; v < DV_; ++v) pg[v]   = fmaf(wg[v * C_ + c], xv, pg[v]);
            }
#pragma unroll
            for (int d = 0; d < D_; ++d)  aphi[d] = fmaxf(aphi[d], pphi[d]);
#pragma unroll
            for (int v = 0; v < DV_; ++v) ag[v]   = fmaxf(ag[v], pg[v]);
        }
    }
#pragma unroll
    for (int d = 0; d < D_; ++d)
        phi[((size_t)b * M_ + pm) * D_ + d] = f2bf(aphi[d]);
#pragma unroll
    for (int v = 0; v < DV_; ++v)
        g[((size_t)b * DV_ + v) * M_ + pm] = f2bf(ag[v]);
}

// ---------------------------------------------------------------------------
// Flash-attention + output projection + residual.
// One wave per 16-query tile; keys processed 32 at a time (full WMMA K).
// C/D tile layout (16x16 f32): vgpr r, lanes 0-15 -> (M=r, N=lane),
//                              lanes 16-31 -> (M=r+8, N=lane-16).
// A operand (16x32 bf16):  lane L holds row M=L%16; K set = L<16 ? {0..7,16..23}
//                          : {8..15,24..31}.
// B operand (32x16 bf16):  lane L holds col N=L%16 with the same K split.
// ---------------------------------------------------------------------------
__global__ void __launch_bounds__(128)
k_attn(const __bf16* __restrict__ theta, const __bf16* __restrict__ phi,
       const __bf16* __restrict__ g, const float* __restrict__ w_o,
       const float* __restrict__ x, const float* __restrict__ gamma,
       float* __restrict__ out) {
    __shared__ float pbuf[4][16 * 32];     // per-wave D-layout <-> A-layout bounce
    __shared__ float scbuf[4][16];         // per-wave row-broadcast buffer

    const int lane = threadIdx.x & 31;
    const int wid  = threadIdx.x >> 5;
    const int tile = blockIdx.x * 4 + wid;       // B * N/16 = 2048 tiles
    const int b    = tile >> 10;
    const int n0   = (tile & 1023) << 4;

    const int col = lane & 15;
    const int hi  = lane >> 4;
    const int kb  = hi ? 8 : 0;

    // ---- Q operand: depth 8 zero-padded into K=32 A-matrix ----
    v16bf aq = {};
    if (!hi) {
        v8bf q = *(const v8bf*)(theta + ((size_t)b * N_ + n0 + col) * D_);
#pragma unroll
        for (int e = 0; e < 8; ++e) aq[e] = q[e];
    }

    // ---- w_o as four B-operand c-tiles (loop invariant) ----
    v16bf bw[4];
#pragma unroll
    for (int ct = 0; ct < 4; ++ct) {
        int c = ct * 16 + col;
#pragma unroll
        for (int e = 0; e < 8; ++e) {
            bw[ct][e]     = f2bf(w_o[c * DV_ + kb + e]);
            bw[ct][8 + e] = f2bf(w_o[c * DV_ + 16 + kb + e]);
        }
    }

    v8f o0 = {}, o1 = {};
    float m_run = -3.0e38f;     // stats for row q = col (replicated in lane pair)
    float l_run = 0.f;

    float* pb = pbuf[wid];
    float* sb = scbuf[wid];

    for (int k0 = 0; k0 < M_; k0 += 32) {
        // ---- prefetch next key tile's streams into cache ----
        if (k0 + 32 < M_) {
            __builtin_prefetch(phi + ((size_t)b * M_ + k0 + 32 + col) * D_, 0, 0);
            __builtin_prefetch(g + ((size_t)b * DV_ + col) * M_ + k0 + 32, 0, 0);
        }

        // ---- key (phi) B operands: two 16-key sub-tiles, depth 8 padded ----
        v16bf bk0 = {}, bk1 = {};
        if (!hi) {
            v8bf p0 = *(const v8bf*)(phi + ((size_t)b * M_ + k0 + col) * D_);
            v8bf p1 = *(const v8bf*)(phi + ((size_t)b * M_ + k0 + 16 + col) * D_);
#pragma unroll
            for (int e = 0; e < 8; ++e) { bk0[e] = p0[e]; bk1[e] = p1[e]; }
        }
        v8f s0 = {}, s1 = {};
        s0 = __builtin_amdgcn_wmma_f32_16x16x32_bf16(false, aq, false, bk0,
                                                     (short)0, s0, false, false);
        s1 = __builtin_amdgcn_wmma_f32_16x16x32_bf16(false, aq, false, bk1,
                                                     (short)0, s1, false, false);

        // ---- transpose raw S to A-layout via LDS ----
#pragma unroll
        for (int r = 0; r < 8; ++r) {
            int q = r + (hi << 3);
            pb[q * 32 + col]      = s0[r];
            pb[q * 32 + 16 + col] = s1[r];
        }
        asm volatile("s_wait_dscnt 0x0" ::: "memory");

        const float* row = pb + col * 32;          // row q = col
        v4f c0 = *(const v4f*)(row + kb);
        v4f c1 = *(const v4f*)(row + kb + 4);
        v4f c2 = *(const v4f*)(row + 16 + kb);
        v4f c3 = *(const v4f*)(row + 16 + kb + 4);
        float vals[16];
#pragma unroll
        for (int e = 0; e < 4; ++e) {
            vals[e]      = c0[e];
            vals[4 + e]  = c1[e];
            vals[8 + e]  = c2[e];
            vals[12 + e] = c3[e];
        }

        // ---- online softmax: local reduce + single pair shuffle ----
        float tmax = vals[0];
#pragma unroll
        for (int e = 1; e < 16; ++e) tmax = fmaxf(tmax, vals[e]);
        tmax = fmaxf(tmax, __shfl_xor(tmax, 16));
        float mn = fmaxf(m_run, tmax);
        float sc = __expf(m_run - mn);
        m_run = mn;

        v16bf ap;                                  // P in A-layout, bf16
        float rs = 0.f;
#pragma unroll
        for (int e = 0; e < 16; ++e) {
            float p = __expf(vals[e] - mn);
            rs += p;
            ap[e] = f2bf(p);
        }
        rs += __shfl_xor(rs, 16);
        l_run = l_run * sc + rs;

        // ---- broadcast sc to D-layout rows and rescale accumulators ----
        if (!hi) sb[col] = sc;
        asm volatile("s_wait_dscnt 0x0" ::: "memory");
        v4f sA = *(const v4f*)(sb + (hi << 3));
        v4f sB = *(const v4f*)(sb + (hi << 3) + 4);
#pragma unroll
        for (int r = 0; r < 4; ++r) {
            o0[r]     *= sA[r];
            o1[r]     *= sA[r];
            o0[4 + r] *= sB[r];
            o1[4 + r] *= sB[r];
        }

        // ---- V operands from v-major g: contiguous 16B chunks per lane ----
        v16bf bv0, bv1;
        {
            const __bf16* g0 = g + ((size_t)b * DV_ + col) * M_ + k0;
            const __bf16* g1 = g + ((size_t)b * DV_ + 16 + col) * M_ + k0;
            v8bf d00 = *(const v8bf*)(g0 + kb);
            v8bf d01 = *(const v8bf*)(g0 + 16 + kb);
            v8bf d10 = *(const v8bf*)(g1 + kb);
            v8bf d11 = *(const v8bf*)(g1 + 16 + kb);
#pragma unroll
            for (int e = 0; e < 8; ++e) {
                bv0[e] = d00[e]; bv0[8 + e] = d01[e];
                bv1[e] = d10[e]; bv1[8 + e] = d11[e];
            }
        }
        o0 = __builtin_amdgcn_wmma_f32_16x16x32_bf16(false, ap, false, bv0,
                                                     (short)0, o0, false, false);
        o1 = __builtin_amdgcn_wmma_f32_16x16x32_bf16(false, ap, false, bv1,
                                                     (short)0, o1, false, false);
    }

    // ---- bounce O to A-layout; normalize with this lane's own 1/l ----
#pragma unroll
    for (int r = 0; r < 8; ++r) {
        int q = r + (hi << 3);
        pb[q * 32 + col]      = o0[r];
        pb[q * 32 + 16 + col] = o1[r];
    }
    asm volatile("s_wait_dscnt 0x0" ::: "memory");
    v16bf ao;
    {
        float invl = 1.0f / l_run;
        const float* row = pb + col * 32;
        v4f c0 = *(const v4f*)(row + kb);
        v4f c1 = *(const v4f*)(row + kb + 4);
        v4f c2 = *(const v4f*)(row + 16 + kb);
        v4f c3 = *(const v4f*)(row + 16 + kb + 4);
#pragma unroll
        for (int e = 0; e < 4; ++e) {
            ao[e]      = f2bf(c0[e] * invl);
            ao[4 + e]  = f2bf(c1[e] * invl);
            ao[8 + e]  = f2bf(c2[e] * invl);
            ao[12 + e] = f2bf(c3[e] * invl);
        }
    }

    // ---- out = x + gamma * (w_o @ O), 4 c-tiles of 16 channels ----
    float gm = gamma[0];
#pragma unroll
    for (int ct = 0; ct < 4; ++ct) {
        v8f dt = {};
        dt = __builtin_amdgcn_wmma_f32_16x16x32_bf16(false, ao, false, bw[ct],
                                                     (short)0, dt, false, false);
        int c = ct * 16 + col;
#pragma unroll
        for (int r = 0; r < 8; ++r) {
            int qrow = n0 + r + (hi << 3);
            size_t idx = ((size_t)b * C_ + c) * N_ + qrow;
            out[idx] = x[idx] + gm * dt[r];
        }
    }
}

// ---------------------------------------------------------------------------
// Host-side launch. Inputs: x, w_theta, w_phi, w_g, w_o, gamma (all f32).
// Workspace: theta bf16 [B,N,8] | phi bf16 [B,M,8] | g bf16 [B,32,M] ~1.13 MB.
// ---------------------------------------------------------------------------
extern "C" void kernel_launch(void* const* d_in, const int* in_sizes, int n_in,
                              void* d_out, int out_size, void* d_ws, size_t ws_size,
                              hipStream_t stream) {
    (void)in_sizes; (void)n_in; (void)out_size; (void)ws_size;

    const float* x       = (const float*)d_in[0];
    const float* w_theta = (const float*)d_in[1];
    const float* w_phi   = (const float*)d_in[2];
    const float* w_g     = (const float*)d_in[3];
    const float* w_o     = (const float*)d_in[4];
    const float* gamma   = (const float*)d_in[5];
    float*       out     = (float*)d_out;

    char* ws = (char*)d_ws;
    __bf16* theta = (__bf16*)(ws);                                   // 512 KB
    __bf16* phi   = (__bf16*)(ws + (size_t)B_ * N_ * D_ * 2);        // 128 KB
    __bf16* g     = (__bf16*)(ws + (size_t)B_ * N_ * D_ * 2
                                 + (size_t)B_ * M_ * D_ * 2);        // 512 KB

    k_theta<<<(B_ * N_) / 256, 256, 0, stream>>>(x, w_theta, theta);
    k_pool<<<(B_ * M_) / 256, 256, 0, stream>>>(x, w_phi, w_g, phi, g);
    k_attn<<<(B_ * (N_ / 16)) / 4, 128, 0, stream>>>(theta, phi, g, w_o, x,
                                                     gamma, out);
}
